// MaskedSelfAttention_71906342469883
// MI455X (gfx1250) — matrix operand order repaired and checked by reading, not compiled, not run on hardware
//
#include <hip/hip_runtime.h>
#include <hip/hip_bf16.h>

typedef _Float16 half_t;
typedef __attribute__((ext_vector_type(16))) _Float16 v16h;
typedef __attribute__((ext_vector_type(8)))  _Float16 v8h;
typedef __attribute__((ext_vector_type(8)))  float    v8f;

#define B_     8
#define N_     1024
#define H_     1024
#define HEADS_ 16
#define DH_    64
#define M_TOK  (B_ * N_)   // 8192 token rows

__device__ __forceinline__ v8f wmma_f16(v16h a, v16h b, v8f c) {
  // D = A(16x32 f16) * B(32x16 f16) + C(16x16 f32)
  return __builtin_amdgcn_wmma_f32_16x16x32_f16(false, a, false, b, (short)0, c, false, false);
}

// async copy 16B/lane from global to LDS (ASYNCcnt-tracked). LDS byte offset is
// the low 32 bits of a flat pointer into shared memory (aperture layout, ISA 10.2).
__device__ __forceinline__ void async_g2lds_b128(unsigned lds_off, uint64_t gaddr) {
  asm volatile("global_load_async_to_lds_b128 %0, %1, off"
               :: "v"(lds_off), "v"(gaddr) : "memory");
}
__device__ __forceinline__ void wait_asynccnt0() {
  asm volatile("s_wait_asynccnt 0x0" ::: "memory");
}

// ---------------------------------------------------------------- conversions
__global__ void cvt_f32_f16(const float* __restrict__ in, half_t* __restrict__ out, int n) {
  int i = blockIdx.x * blockDim.x + threadIdx.x;
  int stride = gridDim.x * blockDim.x;
  for (; i < n; i += stride) out[i] = (half_t)in[i];
}

// ---------------------------------------------------------------- WMMA GEMM
// C[M,N] = A[M,K](f16) @ W[K,N](f16) + bias[N](f32)
// Block tile 128(M) x 128(N), 8 waves; wave w owns rows [w*16, w*16+16) x all 128 cols.
#define GK_BM 128
#define GK_BN 128
#define GK_KC 32

template <bool F16OUT>
__global__ __launch_bounds__(256) void gemm_wmma(const half_t* __restrict__ A,
                                                 const half_t* __restrict__ W,
                                                 const float* __restrict__ bias,
                                                 void* __restrict__ Cout,
                                                 int M, int K, int Nn) {
  __shared__ half_t lA[GK_BM][GK_KC];   // 8 KB, row-major (async-filled)
  __shared__ half_t lBt[GK_BN][GK_KC];  // 8 KB, transposed: [n][k]

  const int tid  = threadIdx.x;
  const int wave = tid >> 5;
  const int lane = tid & 31;
  const int m0 = blockIdx.y * GK_BM;
  const int n0 = blockIdx.x * GK_BN;

  const int arow   = wave * 16 + (lane & 15);  // A fragment row within block
  const int koff8  = (lane >> 4) * 8;          // unused; A fragment also uses koff16
  const int koff16 = (lane >> 4) * 16;         // B fragment k-offset
  const int ncol   = lane & 15;                // B/C fragment column

  // A staging addresses: thread t owns a 32B chunk of the 128x32-half tile
  const int a_r  = tid >> 2;               // tile row        (32B*tid)/64B-per-row... (tid*32)/64
  const int a_ch = (tid & 3) * 8;          // halves within row: 0,8,16,24? -> need 16-half granularity
  // recompute cleanly: flat byte off = tid*32; row = off/64; colhalf = (off%64)/2
  const int fb   = tid * 32;
  const int ar   = fb >> 6;
  const int ac   = (fb & 63) >> 1;
  const unsigned lds_a = (unsigned)(uintptr_t)&lA[ar][ac];

  v8f acc[8] = {};

  for (int k0 = 0; k0 < K; k0 += GK_KC) {
    // --- A block 128x32 halves via async global->LDS (2 x b128 per thread)
    {
      uint64_t g = (uint64_t)(uintptr_t)(A + (size_t)(m0 + ar) * K + k0 + ac);
      async_g2lds_b128(lds_a, g);
      async_g2lds_b128(lds_a + 16, g + 16);
    }
    // --- W block 32x128 halves, staged transposed (16 halves per thread)
    {
      int kk = tid >> 3;
      int c  = (tid & 7) * 16;
      v16h t = *(const v16h*)(W + (size_t)(k0 + kk) * Nn + n0 + c);
#pragma unroll
      for (int i = 0; i < 16; ++i) lBt[c + i][kk] = t[i];
    }
    wait_asynccnt0();
    __syncthreads();

    // A fragment (16x32): lane<16 holds K {0..15}, lane>=16 holds K {16..31}
    v16h af;
    {
      v8h lo = *(const v8h*)&lA[arow][koff16];
      v8h hi = *(const v8h*)&lA[arow][koff16 + 8];
#pragma unroll
      for (int i = 0; i < 8; ++i) { af[i] = lo[i]; af[i + 8] = hi[i]; }
    }
#pragma unroll
    for (int j = 0; j < 8; ++j) {
      v16h bf = *(const v16h*)&lBt[j * 16 + ncol][koff16];  // contiguous 16 halves
      acc[j] = wmma_f16(af, bf, acc[j]);
    }
    __syncthreads();
  }

  // C/D layout: VGPR r -> M = (lane>=16)*8 + r, N = lane&15
  const int crow = m0 + wave * 16 + (lane >> 4) * 8;
#pragma unroll
  for (int j = 0; j < 8; ++j) {
    int col = n0 + j * 16 + ncol;
    float bc = bias[col];
#pragma unroll
    for (int r = 0; r < 8; ++r) {
      float v = acc[j][r] + bc;
      size_t idx = (size_t)(crow + r) * Nn + col;
      if (F16OUT) ((half_t*)Cout)[idx] = (half_t)v;
      else        ((float*)Cout)[idx]  = v;
    }
  }
}

// ---------------------------------------------------------------- attention
// 1 wave per (b, h, 16-query tile). S[16][1024] f32 (64 KB) + Vt (4 KB) in LDS.
__global__ __launch_bounds__(32) void attn_wmma(const half_t* __restrict__ Qh,
                                                const half_t* __restrict__ Kh,
                                                const half_t* __restrict__ Vh,
                                                const int* __restrict__ role_mask,
                                                half_t* __restrict__ Ctx) {
  __shared__ float  S[16][N_];    // scores, then normalized probs (in place)
  __shared__ half_t Vt[DH_][32];  // V chunk transposed: [dh][key_local]

  const int lane = threadIdx.x;
  int idx = blockIdx.x;
  const int qt = idx & 63;         idx >>= 6;
  const int h  = idx & (HEADS_-1); idx >>= 4;
  const int b  = idx;

  const int arow   = lane & 15;
  const int koff8  = (lane >> 4) * 8;
  const int koff16 = (lane >> 4) * 16;
  const int ncol   = lane & 15;
  const size_t hoff = (size_t)h * DH_;

  // ---- Q fragments (2 chunks of K=32 covering DH=64), held in registers
  v16h qf[2];
  {
    const half_t* qb = Qh + ((size_t)(b * N_ + qt * 16 + arow)) * H_ + hoff;
#pragma unroll
    for (int c = 0; c < 2; ++c) {
      v8h lo = *(const v8h*)(qb + c * 32 + koff16);
      v8h hi = *(const v8h*)(qb + c * 32 + koff16 + 8);
#pragma unroll
      for (int i = 0; i < 8; ++i) { qf[c][i] = lo[i]; qf[c][i + 8] = hi[i]; }
    }
  }

  const float scale = 0.125f;  // 1/sqrt(64)

  // ---- scores: S = (Q K^T) * scale, masked
  for (int j = 0; j < 64; ++j) {
    v8f acc = {};
    // B fragment = K^T chunk: column n = key j*16+ncol, k contiguous in dh
    const half_t* kb = Kh + ((size_t)(b * N_ + j * 16 + ncol)) * H_ + hoff;
#pragma unroll
    for (int c = 0; c < 2; ++c) {
      v16h kf = *(const v16h*)(kb + c * 32 + koff16);  // 32B contiguous
      acc = wmma_f16(qf[c], kf, acc);
    }
#pragma unroll
    for (int r = 0; r < 8; ++r) {
      int qrow = (lane >> 4) * 8 + r;
      int key  = j * 16 + ncol;
      int m = role_mask[((size_t)(b * N_) + qt * 16 + qrow) * N_ + key];
      S[qrow][key] = (m == 0) ? -1e9f : acc[r] * scale;
    }
  }
  __syncthreads();

  // ---- softmax: 2 lanes per row (lane & lane+16), 512 cols each
  {
    const int srow = lane & 15;
    const int c0 = (lane >> 4) * 512;
    float mx = -1e30f;
    for (int c = c0; c < c0 + 512; ++c) mx = fmaxf(mx, S[srow][c]);
    mx = fmaxf(mx, __shfl_xor(mx, 16, 32));
    float sum = 0.f;
    for (int c = c0; c < c0 + 512; ++c) {
      float e = __expf(S[srow][c] - mx);
      sum += e;
      S[srow][c] = e;
    }
    sum += __shfl_xor(sum, 16, 32);
    float inv = 1.f / sum;
    for (int c = c0; c < c0 + 512; ++c) S[srow][c] *= inv;
  }
  __syncthreads();

  // ---- ctx = P(16x1024) @ V(1024x64), 4 N-tiles of 16
  v8f cacc[4] = {};
  for (int kk = 0; kk < 32; ++kk) {
    // stage V chunk [32 keys][64 dh] transposed into Vt (one key row per lane)
    {
      const half_t* vb = Vh + ((size_t)(b * N_) + kk * 32 + lane) * H_ + hoff;
#pragma unroll
      for (int p = 0; p < 4; ++p) {
        v16h t = *(const v16h*)(vb + p * 16);  // 32B contiguous per lane
#pragma unroll
        for (int i = 0; i < 16; ++i) Vt[p * 16 + i][lane] = t[i];
      }
    }
    __syncthreads();

    v16h pf;
#pragma unroll
    for (int i = 0; i < 16; ++i) {
      int kloc = i + koff16;
      pf[i] = (half_t)S[arow][kk * 32 + kloc];
    }
#pragma unroll
    for (int j = 0; j < 4; ++j) {
      v16h vf = *(const v16h*)&Vt[j * 16 + ncol][koff16];  // contiguous 16 halves
      cacc[j] = wmma_f16(pf, vf, cacc[j]);
    }
    __syncthreads();
  }

  // ---- write ctx (token-major [B*N][H], head-interleaved => matches reshape)
#pragma unroll
  for (int j = 0; j < 4; ++j) {
#pragma unroll
    for (int r = 0; r < 8; ++r) {
      int qrow = qt * 16 + (lane >> 4) * 8 + r;
      Ctx[((size_t)(b * N_ + qrow)) * H_ + hoff + j * 16 + ncol] = (half_t)cacc[j][r];
    }
  }
}

// ---------------------------------------------------------------- residual + LayerNorm
__global__ __launch_bounds__(256) void ln_kernel(const float* __restrict__ P32,
                                                 const float* __restrict__ X,
                                                 const float* __restrict__ gamma,
                                                 const float* __restrict__ beta,
                                                 float* __restrict__ out) {
  __shared__ float red[16];
  const int row = blockIdx.x;
  const int tid = threadIdx.x;

  float y[4], s = 0.f, ss = 0.f;
#pragma unroll
  for (int i = 0; i < 4; ++i) {
    int c = tid + i * 256;
    float v = P32[(size_t)row * H_ + c] + X[(size_t)row * H_ + c];
    y[i] = v; s += v; ss += v * v;
  }
#pragma unroll
  for (int o = 16; o > 0; o >>= 1) { s += __shfl_xor(s, o, 32); ss += __shfl_xor(ss, o, 32); }
  if ((tid & 31) == 0) { red[tid >> 5] = s; red[8 + (tid >> 5)] = ss; }
  __syncthreads();
  float ts = 0.f, tss = 0.f;
#pragma unroll
  for (int w = 0; w < 8; ++w) { ts += red[w]; tss += red[8 + w]; }
  float mu  = ts * (1.f / H_);
  float var = tss * (1.f / H_) - mu * mu;
  float inv = rsqrtf(var + 1e-12f);
#pragma unroll
  for (int i = 0; i < 4; ++i) {
    int c = tid + i * 256;
    out[(size_t)row * H_ + c] = (y[i] - mu) * inv * gamma[c] + beta[c];
  }
}

// ---------------------------------------------------------------- row gather
__global__ void gather_rows(const float* __restrict__ ctx, const int* __restrict__ conv_len,
                            float* __restrict__ emo, float* __restrict__ persp) {
  int b = blockIdx.x;
  int c = threadIdx.x;
  int base = 6 * conv_len[b];
  persp[(size_t)b * H_ + c] = ctx[((size_t)(b * N_) + base + 2) * H_ + c];
  emo[(size_t)b * H_ + c]   = ctx[((size_t)(b * N_) + base + 3) * H_ + c];
}

// ---------------------------------------------------------------- launcher
extern "C" void kernel_launch(void* const* d_in, const int* in_sizes, int n_in,
                              void* d_out, int out_size, void* d_ws, size_t ws_size,
                              hipStream_t stream) {
  const float* X     = (const float*)d_in[0];
  const int*   mask  = (const int*)d_in[1];
  const int*   clen  = (const int*)d_in[2];
  const float* Wq = (const float*)d_in[3];  const float* bq = (const float*)d_in[4];
  const float* Wk = (const float*)d_in[5];  const float* bk = (const float*)d_in[6];
  const float* Wv = (const float*)d_in[7];  const float* bv = (const float*)d_in[8];
  const float* Wo = (const float*)d_in[9];  const float* bo = (const float*)d_in[10];
  const float* gamma = (const float*)d_in[11];
  const float* beta  = (const float*)d_in[12];

  float* out_ctx   = (float*)d_out;                 // [B,N,H]
  float* out_emo   = out_ctx + (size_t)M_TOK * H_;  // [B,H]
  float* out_persp = out_emo + (size_t)B_ * H_;     // [B,H]

  const size_t MB = 1u << 20;
  char* ws = (char*)d_ws;
  half_t* Xh  = (half_t*)(ws + 0 * MB);    // 16 MB; reused as Ctxh after QKV GEMMs
  half_t* Wqh = (half_t*)(ws + 16 * MB);   // 2 MB each
  half_t* Wkh = (half_t*)(ws + 18 * MB);
  half_t* Wvh = (half_t*)(ws + 20 * MB);
  half_t* Woh = (half_t*)(ws + 22 * MB);
  half_t* Qh  = (half_t*)(ws + 24 * MB);   // 16 MB
  half_t* Kh  = (half_t*)(ws + 40 * MB);   // 16 MB
  half_t* Vh  = (half_t*)(ws + 56 * MB);   // 16 MB
  float*  P32 = (float*)(ws + 24 * MB);    // 32 MB, overlaps Qh/Kh (dead after attn)
  half_t* Ctxh = Xh;                       // overlaps Xh (dead after QKV GEMMs)

  const int HH = H_ * H_;
  cvt_f32_f16<<<4096, 256, 0, stream>>>(X,  Xh,  M_TOK * H_);
  cvt_f32_f16<<<1024, 256, 0, stream>>>(Wq, Wqh, HH);
  cvt_f32_f16<<<1024, 256, 0, stream>>>(Wk, Wkh, HH);
  cvt_f32_f16<<<1024, 256, 0, stream>>>(Wv, Wvh, HH);
  cvt_f32_f16<<<1024, 256, 0, stream>>>(Wo, Woh, HH);

  dim3 ggrid(H_ / GK_BN, M_TOK / GK_BM);  // (8, 64)
  gemm_wmma<true><<<ggrid, 256, 0, stream>>>(Xh, Wqh, bq, Qh, M_TOK, H_, H_);
  gemm_wmma<true><<<ggrid, 256, 0, stream>>>(Xh, Wkh, bk, Kh, M_TOK, H_, H_);
  gemm_wmma<true><<<ggrid, 256, 0, stream>>>(Xh, Wvh, bv, Vh, M_TOK, H_, H_);

  attn_wmma<<<B_ * HEADS_ * (N_ / 16), 32, 0, stream>>>(Qh, Kh, Vh, mask, Ctxh);

  gemm_wmma<false><<<ggrid, 256, 0, stream>>>(Ctxh, Woh, bo, P32, M_TOK, H_, H_);

  ln_kernel<<<M_TOK, 256, 0, stream>>>(P32, X, gamma, beta, out_ctx);
  gather_rows<<<B_, H_, 0, stream>>>(out_ctx, clen, out_emo, out_persp);
}
